// AttentionModule_25864293057187
// MI455X (gfx1250) — matrix operand order repaired and checked by reading, not compiled, not run on hardware
//
#include <hip/hip_runtime.h>
#include <hip/hip_bf16.h>
#include <math.h>

typedef _Float16 h16;
typedef _Float16 v16h __attribute__((ext_vector_type(16)));
typedef _Float16 v8h  __attribute__((ext_vector_type(8)));
typedef float    v8f  __attribute__((ext_vector_type(8)));

#define NBATCH 2
#define NC     64
#define NH     96
#define NW     96
#define HD     48
#define NL     2304   // 48*48
#define NKC    576    // 64*9
#define ATT_SCALE 10.0f

// ---------------- WMMA fragment loaders (CDNA5 wave32 layouts) ----------------
// A (16x32 f16): lane<16: M=lane, K = {0..7, 16..23}; lane>=16: M=lane-16, K = {8..15, 24..31}
__device__ __forceinline__ v16h frag_a(const h16* Xs, int row_base, int lane) {
  int r = row_base + (lane & 15);
  int klo = (lane < 16) ? 0 : 8;
  const h16* p = Xs + r * 32;
  v16h f;
  *(v8h*)&f       = *(const v8h*)(p + klo);
  *((v8h*)&f + 1) = *(const v8h*)(p + klo + 16);
  return f;
}
// B (32x16 f16), staged transposed as Bt[n][k]: lane<16: N=lane, K = {0..7, 16..23}; lane>=16: N=lane-16, K = {8..15, 24..31}
__device__ __forceinline__ v16h frag_b(const h16* Xs, int row_base, int lane) {
  int n = row_base + (lane & 15);
  int klo = (lane < 16) ? 0 : 8;
  const h16* p = Xs + n * 32 + klo;
  v16h f;
  *(v8h*)&f       = *(const v8h*)(p);
  *((v8h*)&f + 1) = *(const v8h*)(p + 16);
  return f;
}

// ---------------- utility ----------------
__global__ void k_zero(unsigned int* p, long n) {
  long i = (long)blockIdx.x * blockDim.x + threadIdx.x;
  long st = (long)gridDim.x * blockDim.x;
  for (; i < n; i += st) p[i] = 0u;
}

// ---------------- prep: im2col of f_ds -> Fmat16[s][p][k], k=c*9+t ----------------
__global__ void k_prep_fmat(const float* __restrict__ f, h16* __restrict__ Fmat) {
  long total = (long)NBATCH * NL * NKC;
  long i = (long)blockIdx.x * blockDim.x + threadIdx.x;
  long st = (long)gridDim.x * blockDim.x;
  for (; i < total; i += st) {
    int s = (int)(i / ((long)NL * NKC));
    long r = i % ((long)NL * NKC);
    int p = (int)(r / NKC), k = (int)(r % NKC);
    int c = k / 9, t = k % 9, ti = t / 3, tj = t % 3;
    int py = p / HD + ti - 1, px = p % HD + tj - 1;
    float v = 0.f;
    if (py >= 0 && py < HD && px >= 0 && px < HD)
      v = f[(((long)s * NC + c) * NH + 2 * py) * NW + 2 * px];
    Fmat[i] = (h16)v;
  }
}

// ---------------- prep: per-(c0,pos) patch sum-of-squares of b_ds ----------------
__global__ void k_prep_psum(const float* __restrict__ b, float* __restrict__ psum) {
  long total = (long)NBATCH * NC * NL;
  long i = (long)blockIdx.x * blockDim.x + threadIdx.x;
  long st = (long)gridDim.x * blockDim.x;
  for (; i < total; i += st) {
    int s = (int)(i / ((long)NC * NL));
    long flat = i % ((long)NC * NL);
    int c0 = (int)(flat / NL), pos = (int)(flat % NL);
    int posy = pos / HD, posx = pos % HD;
    float acc = 0.f;
    for (int ti = 0; ti < 3; ti++)
      for (int tj = 0; tj < 3; tj++) {
        int y = posy + ti - 1, x = posx + tj - 1;
        if (y >= 0 && y < HD && x >= 0 && x < HD) {
          float v = b[(((long)s * NC + c0) * NH + 2 * y) * NW + 2 * x];
          acc += v * v;
        }
      }
    psum[i] = acc;
  }
}

// norm over scrambled axis: norm[c]^2 = sum_x psum[x*64+c]  (flat = x*64+c = c0*2304+pos)
__global__ void k_prep_rnorm(const float* __restrict__ psum, float* __restrict__ rnorm) {
  int i = blockIdx.x * blockDim.x + threadIdx.x;
  if (i >= NBATCH * NC) return;
  int s = i / NC, c = i % NC;
  const float* ps = psum + (long)s * NC * NL;
  float acc = 0.f;
  for (int x = 0; x < NL; x++) acc += ps[(long)x * NC + c];
  float n = fmaxf(sqrtf(acc), 1e-4f);
  rnorm[i] = 1.f / n;
}

// Wmat16[s][o][k]: W[o,c,i,j]/norm[c], with scrambled (c0,pos) = divmod(o*64+c, 2304)
__global__ void k_prep_wmat(const float* __restrict__ b, const float* __restrict__ rnorm,
                            h16* __restrict__ Wmat) {
  long total = (long)NBATCH * NL * NKC;
  long i = (long)blockIdx.x * blockDim.x + threadIdx.x;
  long st = (long)gridDim.x * blockDim.x;
  for (; i < total; i += st) {
    int s = (int)(i / ((long)NL * NKC));
    long r = i % ((long)NL * NKC);
    int o = (int)(r / NKC), k = (int)(r % NKC);
    int c = k / 9, t = k % 9, ti = t / 3, tj = t % 3;
    long flat = (long)o * NC + c;
    int c0 = (int)(flat / NL), pos = (int)(flat % NL);
    int py = pos / HD + ti - 1, px = pos % HD + tj - 1;
    float v = 0.f;
    if (py >= 0 && py < HD && px >= 0 && px < HD)
      v = b[(((long)s * NC + c0) * NH + 2 * py) * NW + 2 * px];
    Wmat[i] = (h16)(v * rnorm[s * NC + c]);
  }
}

// mm[s][q] = mean of 3x3 (zero-padded) neighborhood of m_ds, /9
__global__ void k_prep_mm(const float* __restrict__ mask, float* __restrict__ mm) {
  int i = blockIdx.x * blockDim.x + threadIdx.x;
  if (i >= NBATCH * NL) return;
  int s = i / NL, q = i % NL;
  int qy = q / HD, qx = q % HD;
  float acc = 0.f;
  for (int ti = 0; ti < 3; ti++)
    for (int tj = 0; tj < 3; tj++) {
      int y = qy + ti - 1, x = qx + tj - 1;
      if (y >= 0 && y < HD && x >= 0 && x < HD)
        acc += mask[((long)s * NH + 2 * y) * NW + 2 * x];
    }
  mm[i] = acc / 9.f;
}

// deconv weights: RW[s][t=ky*4+kx][c][x] = b[c0, 2*posy+2-ky, 2*posx+2-kx] (scrambled c0,pos)
__global__ void k_prep_rw(const float* __restrict__ b, h16* __restrict__ RW) {
  long total = (long)NBATCH * 16 * NC * NL;
  long i = (long)blockIdx.x * blockDim.x + threadIdx.x;
  long st = (long)gridDim.x * blockDim.x;
  for (; i < total; i += st) {
    int s = (int)(i / (16L * NC * NL));
    long r = i % (16L * NC * NL);
    int t = (int)(r / ((long)NC * NL));
    long rr = r % ((long)NC * NL);
    int c = (int)(rr / NL), x = (int)(rr % NL);
    int ky = t / 4, kx = t % 4;
    long flat = (long)x * NC + c;
    int c0 = (int)(flat / NL), pos = (int)(flat % NL);
    int yy = 2 * (pos / HD) + 2 - ky, xx = 2 * (pos % HD) + 2 - kx;
    float v = 0.f;
    if (yy >= 0 && yy < NH && xx >= 0 && xx < NW)
      v = b[(((long)s * NC + c0) * NH + yy) * NW + xx];
    RW[i] = (h16)v;
  }
}

// conv weights transposed+f16: WT[cout][t*64+cin]
__global__ void k_prep_wconv(const float* __restrict__ w1, const float* __restrict__ w2,
                             h16* __restrict__ W1T, h16* __restrict__ W2T) {
  int i = blockIdx.x * blockDim.x + threadIdx.x;
  if (i >= NC * NKC) return;
  int co = i / NKC, k = i % NKC;
  int t = k / NC, cin = k % NC;
  int ty = t / 3, tx = t % 3;
  long src = (((long)co * NC + cin) * 3 + ty) * 3 + tx;
  W1T[i] = (h16)w1[src];
  W2T[i] = (h16)w2[src];
}

// ---------------- correlation GEMM: Y[o][p] = Wmat[o,:] . Fmat[p,:] ----------------
// Double-buffered: next K-stage streamed global->LDS with ASYNC copies while WMMAs run.
__global__ void __launch_bounds__(128) k_corr(const h16* __restrict__ Wmat,
                                              const h16* __restrict__ Fmat,
                                              float* __restrict__ Yn) {
  int bid = blockIdx.x;
  int s = bid / (36 * 36);
  int r = bid % (36 * 36);
  int o0 = (r / 36) * 64, p0 = (r % 36) * 64;
  const h16* A = Wmat + (long)s * NL * NKC;
  const h16* Bm = Fmat + (long)s * NL * NKC;
  __shared__ alignas(16) h16 As[2][64 * 32];
  __shared__ alignas(16) h16 Bs[2][64 * 32];
  int tid = threadIdx.x, wv = tid >> 5, lane = tid & 31;
  v8f acc[4] = {};

  // Each stage: 512 16-byte chunks, 128 threads -> exactly 4 async issues per wave.
  auto issue_stage = [&](int k0, int buf) {
#pragma unroll
    for (int c = 0; c < 4; c++) {
      int idx = tid + 128 * c;
      int tile = idx >> 8;               // uniform per iteration
      int row = (idx & 255) >> 2;
      int ch = idx & 3;
      const h16* src = (tile ? Bm + (long)(p0 + row) * NKC : A + (long)(o0 + row) * NKC)
                       + k0 + ch * 8;
      h16* dst = (tile ? Bs[buf] : As[buf]) + row * 32 + ch * 8;
      unsigned ldsa = (unsigned)(unsigned long long)(void*)dst;  // low 32b = LDS offset
      unsigned long long ga = (unsigned long long)src;
      asm volatile("global_load_async_to_lds_b128 %0, %1, off"
                   :: "v"(ldsa), "v"(ga) : "memory");
    }
  };

  issue_stage(0, 0);
  for (int k0 = 0; k0 < NKC; k0 += 32) {
    int buf = (k0 >> 5) & 1;
    if (k0 + 32 < NKC) {
      issue_stage(k0 + 32, buf ^ 1);
      // async loads complete in order: <=4 outstanding means previous stage landed
      asm volatile("s_wait_asynccnt 0x4" ::: "memory");
    } else {
      asm volatile("s_wait_asynccnt 0x0" ::: "memory");
    }
    __syncthreads();
    v16h af = frag_a(As[buf], wv * 16, lane);
#pragma unroll
    for (int n = 0; n < 4; n++) {
      v16h bf = frag_b(Bs[buf], n * 16, lane);
      acc[n] = __builtin_amdgcn_wmma_f32_16x16x32_f16(false, af, false, bf,
                                                      (short)0, acc[n], false, false);
    }
    __syncthreads();
  }
  int nlo = lane & 15;
  int mbase = o0 + wv * 16 + ((lane < 16) ? 0 : 8);
  float* Yr = Yn + (long)s * NL * NL;
#pragma unroll
  for (int n = 0; n < 4; n++) {
    int col = p0 + n * 16 + nlo;
#pragma unroll
    for (int r8 = 0; r8 < 8; r8++)
      Yr[(long)(mbase + r8) * NL + col] = acc[n][r8];
  }
}

// ---------------- fuse pass 1: diagonal 3-sum over flat (o,p) ----------------
__global__ void k_fuse1(const float* __restrict__ Yn, float* __restrict__ Zf) {
  long total = (long)NBATCH * NL * NL;
  long i = (long)blockIdx.x * blockDim.x + threadIdx.x;
  long st = (long)gridDim.x * blockDim.x;
  for (; i < total; i += st) {
    int s = (int)(i / ((long)NL * NL));
    long r = i % ((long)NL * NL);
    long o = r / NL, p = r % NL;
    const float* Ys = Yn + (long)s * NL * NL;
    float acc = 0.f;
    for (int d = -1; d <= 1; d++) {
      long o2 = o + d, p2 = p + d;
      if (o2 >= 0 && o2 < NL && p2 >= 0 && p2 < NL) acc += Ys[o2 * NL + p2];
    }
    Zf[i] = acc;
  }
}

// ---------------- fuse pass 2 on transposed flats; writes Y2T[p][o] ----------------
__global__ void k_fuse2(const float* __restrict__ Zf, float* __restrict__ Y2T) {
  long total = (long)NBATCH * NL * NL;
  long i = (long)blockIdx.x * blockDim.x + threadIdx.x;
  long st = (long)gridDim.x * blockDim.x;
  for (; i < total; i += st) {
    int s = (int)(i / ((long)NL * NL));
    long r = i % ((long)NL * NL);
    int p = (int)(r / NL), o = (int)(r % NL);
    int oh = o / HD, ow = o % HD, ph = p / HD, pw = p % HD;
    int sp = ow * HD + oh, lp = pw * HD + ph;
    const float* Zs = Zf + (long)s * NL * NL;
    float acc = 0.f;
    for (int d = -1; d <= 1; d++) {
      int s2 = sp + d, l2 = lp + d;
      if (s2 >= 0 && s2 < NL && l2 >= 0 && l2 < NL) {
        int o2 = (s2 % HD) * HD + s2 / HD;
        int p2 = (l2 % HD) * HD + l2 / HD;
        acc += Zs[(long)o2 * NL + p2];
      }
    }
    Y2T[i] = acc;
  }
}

// ---------------- mask*scale -> softmax over o -> *mask, argmax, f16 attention ----------------
__global__ void __launch_bounds__(256) k_softmax(const float* __restrict__ Y2T,
                                                 const float* __restrict__ mm,
                                                 h16* __restrict__ Att16,
                                                 int* __restrict__ raw_off) {
  int bid = blockIdx.x;
  int s = bid / NL, p = bid % NL;
  const float* row = Y2T + ((long)s * NL + p) * NL;
  const float* mms = mm + (long)s * NL;
  __shared__ float red[256];
  __shared__ int redi[256];
  int tid = threadIdx.x;
  float mx = -1e30f;
  for (int o = tid; o < NL; o += 256) mx = fmaxf(mx, row[o] * mms[o] * ATT_SCALE);
  red[tid] = mx; __syncthreads();
  for (int st2 = 128; st2; st2 >>= 1) { if (tid < st2) red[tid] = fmaxf(red[tid], red[tid + st2]); __syncthreads(); }
  mx = red[0]; __syncthreads();
  float sum = 0.f;
  for (int o = tid; o < NL; o += 256) sum += __expf(row[o] * mms[o] * ATT_SCALE - mx);
  red[tid] = sum; __syncthreads();
  for (int st2 = 128; st2; st2 >>= 1) { if (tid < st2) red[tid] += red[tid + st2]; __syncthreads(); }
  float inv = 1.f / red[0]; __syncthreads();
  float best = -1e30f; int bi = NL;
  h16* arow = Att16 + ((long)s * NL + p) * NL;
  for (int o = tid; o < NL; o += 256) {
    float a = __expf(row[o] * mms[o] * ATT_SCALE - mx) * inv * mms[o];
    arow[o] = (h16)a;
    if (a > best || (a == best && o < bi)) { best = a; bi = o; }
  }
  red[tid] = best; redi[tid] = bi; __syncthreads();
  for (int st2 = 128; st2; st2 >>= 1) {
    if (tid < st2) {
      if (red[tid + st2] > red[tid] || (red[tid + st2] == red[tid] && redi[tid + st2] < redi[tid])) {
        red[tid] = red[tid + st2]; redi[tid] = redi[tid + st2];
      }
    }
    __syncthreads();
  }
  if (tid == 0) {
    int o = redi[0];
    raw_off[((long)s * NL + p) * 2 + 0] = o / HD;
    raw_off[((long)s * NL + p) * 2 + 1] = o % HD;
  }
}

// offsets: emulate the (48,48,2)->(2,48,48) flat reinterpret, minus coordinate grids
__global__ void k_offsets(const int* __restrict__ raw_off, int* __restrict__ offs_out) {
  int i = blockIdx.x * blockDim.x + threadIdx.x;
  if (i >= NBATCH * 2 * HD * HD) return;
  int s = i / (2 * HD * HD);
  int r = i % (2 * HD * HD);
  int c2 = r / (HD * HD), h2 = (r / HD) % HD, w2 = r % HD;
  int flat = c2 * (HD * HD) + h2 * HD + w2;
  int ph = flat / (2 * HD), rem = flat % (2 * HD);
  int pw = rem / 2, comp = rem % 2;
  int v = raw_off[((long)s * NL + ph * HD + pw) * 2 + comp];
  offs_out[i] = v - ((c2 == 0) ? h2 : w2);
}

// ---------------- paste (transposed conv) as 4-parity x 4-tap WMMA GEMM ----------------
__global__ void __launch_bounds__(128) k_paste(const h16* __restrict__ Att16,
                                               const h16* __restrict__ RW,
                                               h16* __restrict__ yp16) {
  int bid = blockIdx.x;
  int s = bid / (4 * 144);
  int r = bid % (4 * 144);
  int par = r / 144, pt = r % 144;
  int Ypar = par >> 1, Xpar = par & 1;
  int tid = threadIdx.x, wv = tid >> 5, lane = tid & 31;
  const h16* A = Att16 + (long)s * NL * NL;
  __shared__ alignas(16) h16 As[16 * 32];
  __shared__ alignas(16) h16 Bs[64 * 32];
  v8f acc = {};
  for (int j = 0; j < 4; j++) {
    int kyi = j >> 1, kxi = j & 1;
    int ky = Ypar ? (kyi ? 3 : 1) : (kyi ? 2 : 0);
    int dy = Ypar ? (kyi ? 1 : 0) : (kyi ? 0 : -1);
    int kx = Xpar ? (kxi ? 3 : 1) : (kxi ? 2 : 0);
    int dx = Xpar ? (kxi ? 1 : 0) : (kxi ? 0 : -1);
    const h16* Bsrc = RW + ((long)s * 16 + (ky * 4 + kx)) * NC * NL;  // [c][x]
    for (int x0 = 0; x0 < NL; x0 += 32) {
      for (int idx = tid; idx < 320; idx += 128) {
        if (idx < 64) {
          int row = idx >> 2, ch = idx & 3;
          int pp = pt * 16 + row;
          int py = pp / HD + dy, px = pp % HD + dx;
          v8h v = {};
          if (py >= 0 && py < HD && px >= 0 && px < HD)
            v = *(const v8h*)(A + (long)(py * HD + px) * NL + x0 + ch * 8);
          *(v8h*)(As + row * 32 + ch * 8) = v;
        } else {
          int id2 = idx - 64;
          int row = id2 >> 2, ch = id2 & 3;
          const h16* src = Bsrc + (long)row * NL + x0 + ch * 8;
          __builtin_prefetch(src + 32, 0, 1);
          *(v8h*)(Bs + row * 32 + ch * 8) = *(const v8h*)src;
        }
      }
      __syncthreads();
      v16h af = frag_a(As, 0, lane);
      v16h bf = frag_b(Bs, wv * 16, lane);
      acc = __builtin_amdgcn_wmma_f32_16x16x32_f16(false, af, false, bf,
                                                   (short)0, acc, false, false);
      __syncthreads();
    }
  }
  int cch = wv * 16 + (lane & 15);
  int mb = (lane < 16) ? 0 : 8;
#pragma unroll
  for (int r8 = 0; r8 < 8; r8++) {
    int pp = pt * 16 + mb + r8;
    int py = pp / HD, px = pp % HD;
    int Y = 2 * py + Ypar, X = 2 * px + Xpar;
    yp16[(((long)s * 98 + (Y + 1)) * 98 + (X + 1)) * NC + cch] = (h16)(acc[r8] * 0.25f);
  }
}

// ---------------- 3x3 conv as implicit-GEMM WMMA; mode0 -> padded f16, mode1 -> f32 NCHW ----------------
__global__ void __launch_bounds__(128) k_conv(const h16* __restrict__ in16,
                                              const h16* __restrict__ WT,
                                              const float* __restrict__ bias,
                                              h16* __restrict__ z16,
                                              float* __restrict__ outp, int mode) {
  int bid = blockIdx.x;
  int s = bid / 576, pt = bid % 576;
  int tid = threadIdx.x, wv = tid >> 5, lane = tid & 31;
  __shared__ alignas(16) h16 As[16 * 32];
  __shared__ alignas(16) h16 Bs[64 * 32];
  v8f acc = {};
  for (int t = 0; t < 9; t++) {
    int ty = t / 3, tx = t % 3;
    for (int cb = 0; cb < NC; cb += 32) {
      int kk = t * NC + cb;
      for (int idx = tid; idx < 320; idx += 128) {
        if (idx < 64) {
          int row = idx >> 2, ch = idx & 3;
          int pixel = pt * 16 + row;
          int py = pixel / NW + ty, px = pixel % NW + tx;  // padded coords [0,97]
          *(v8h*)(As + row * 32 + ch * 8) =
              *(const v8h*)(in16 + (((long)s * 98 + py) * 98 + px) * NC + cb + ch * 8);
        } else {
          int id2 = idx - 64;
          int row = id2 >> 2, ch = id2 & 3;
          *(v8h*)(Bs + row * 32 + ch * 8) = *(const v8h*)(WT + (long)row * NKC + kk + ch * 8);
        }
      }
      __syncthreads();
      v16h af = frag_a(As, 0, lane);
      v16h bf = frag_b(Bs, wv * 16, lane);
      acc = __builtin_amdgcn_wmma_f32_16x16x32_f16(false, af, false, bf,
                                                   (short)0, acc, false, false);
      __syncthreads();
    }
  }
  int cch = wv * 16 + (lane & 15);
  int mb = (lane < 16) ? 0 : 8;
  float bv = bias[cch];
#pragma unroll
  for (int r8 = 0; r8 < 8; r8++) {
    int pixel = pt * 16 + mb + r8;
    int py = pixel / NW, px = pixel % NW;
    float v = acc[r8] + bv;
    if (mode == 0)
      z16[(((long)s * 98 + py + 1) * 98 + px + 1) * NC + cch] = (h16)v;
    else
      outp[(((long)s * NC + cch) * NH + py) * NW + px] = v;
  }
}

// ---------------- host launch ----------------
extern "C" void kernel_launch(void* const* d_in, const int* in_sizes, int n_in,
                              void* d_out, int out_size, void* d_ws, size_t ws_size,
                              hipStream_t stream) {
  (void)in_sizes; (void)n_in; (void)out_size; (void)ws_size;
  const float* f    = (const float*)d_in[0];
  const float* b    = (const float*)d_in[1];
  const float* mask = (const float*)d_in[2];
  const float* w1   = (const float*)d_in[3];
  const float* b1   = (const float*)d_in[4];
  const float* w2   = (const float*)d_in[5];
  const float* b2   = (const float*)d_in[6];

  size_t off = 0;
  auto take = [&](size_t bytes) { size_t o = off; off += (bytes + 255) & ~(size_t)255; return o; };
  size_t oF   = take(2ull * NL * NKC * 2);            // Fmat16
  size_t oWm  = take(2ull * NL * NKC * 2);            // Wmat16
  size_t oPs  = take(2ull * NC * NL * 4);             // psum
  size_t oRn  = take(2ull * NC * 4);                  // rnorm
  size_t oMm  = take(2ull * NL * 4);                  // mm
  size_t oY   = take(2ull * NL * NL * 4);             // Yn (reused as Y2T)
  size_t oZ   = take(2ull * NL * NL * 4);             // fuse scratch
  size_t oAt  = take(2ull * NL * NL * 2);             // attention f16 [p][o]
  size_t oRW  = take(2ull * 16 * NC * NL * 2);        // deconv weights f16
  size_t oYp  = take(2ull * 98 * 98 * NC * 2);        // padded deconv out f16
  size_t oZ16 = take(2ull * 98 * 98 * NC * 2);        // padded conv1 out f16
  size_t oW1  = take((size_t)NC * NKC * 2);
  size_t oW2  = take((size_t)NC * NKC * 2);
  size_t oRo  = take(2ull * NL * 2 * 4);              // raw argmax (oh,ow) per p

  char* ws = (char*)d_ws;
  h16*   Fmat = (h16*)(ws + oF);
  h16*   Wmat = (h16*)(ws + oWm);
  float* psum = (float*)(ws + oPs);
  float* rnorm= (float*)(ws + oRn);
  float* mm   = (float*)(ws + oMm);
  float* Yn   = (float*)(ws + oY);
  float* Zf   = (float*)(ws + oZ);
  h16*   Att  = (h16*)(ws + oAt);
  h16*   RW   = (h16*)(ws + oRW);
  h16*   yp16 = (h16*)(ws + oYp);
  h16*   z16  = (h16*)(ws + oZ16);
  h16*   W1T  = (h16*)(ws + oW1);
  h16*   W2T  = (h16*)(ws + oW2);
  int*   rawo = (int*)(ws + oRo);

  float* outp = (float*)d_out;
  int*   offs = (int*)(outp + (size_t)NBATCH * NC * NH * NW);

  // zero the padded f16 buffers (borders must be 0 for implicit-GEMM conv)
  k_zero<<<1024, 256, 0, stream>>>((unsigned int*)yp16, (long)(2ull * 98 * 98 * NC * 2) / 4);
  k_zero<<<1024, 256, 0, stream>>>((unsigned int*)z16, (long)(2ull * 98 * 98 * NC * 2) / 4);

  long nFm = (long)NBATCH * NL * NKC;
  k_prep_fmat<<<(int)((nFm + 255) / 256), 256, 0, stream>>>(f, Fmat);
  long nPs = (long)NBATCH * NC * NL;
  k_prep_psum<<<(int)((nPs + 255) / 256), 256, 0, stream>>>(b, psum);
  k_prep_rnorm<<<1, 128, 0, stream>>>(psum, rnorm);
  k_prep_wmat<<<(int)((nFm + 255) / 256), 256, 0, stream>>>(b, rnorm, Wmat);
  k_prep_mm<<<(NBATCH * NL + 255) / 256, 256, 0, stream>>>(mask, mm);
  long nRW = (long)NBATCH * 16 * NC * NL;
  k_prep_rw<<<(int)((nRW + 255) / 256), 256, 0, stream>>>(b, RW);
  k_prep_wconv<<<(NC * NKC + 255) / 256, 256, 0, stream>>>(w1, w2, W1T, W2T);

  k_corr<<<NBATCH * 36 * 36, 128, 0, stream>>>(Wmat, Fmat, Yn);
  k_fuse1<<<4096, 256, 0, stream>>>(Yn, Zf);
  k_fuse2<<<4096, 256, 0, stream>>>(Zf, Yn);  // Yn now holds Y2T[p][o]
  k_softmax<<<NBATCH * NL, 256, 0, stream>>>(Yn, mm, Att, rawo);
  k_offsets<<<(NBATCH * 2 * HD * HD + 255) / 256, 256, 0, stream>>>(rawo, offs);
  k_paste<<<NBATCH * 4 * 144, 128, 0, stream>>>(Att, RW, yp16);
  k_conv<<<NBATCH * 576, 128, 0, stream>>>(yp16, W1T, b1, z16, outp, 0);
  k_conv<<<NBATCH * 576, 128, 0, stream>>>(z16, W2T, b2, z16, outp, 1);
}
